// SphericalGaborLayer_10488310137366
// MI455X (gfx1250) — matrix operand order repaired and checked by reading, not compiled, not run on hardware
//
#include <hip/hip_runtime.h>
#include <hip/hip_bf16.h>

typedef __attribute__((ext_vector_type(16))) _Float16 v16h;
typedef __attribute__((ext_vector_type(8)))  float    v8f;
typedef __attribute__((ext_vector_type(4)))  float    v4f;

#define OMEGA_0   30.0f
#define SIGMA_0   10.0f
#define NPTS      262144
#define WD        128
#define OUTF      256
#define PI_F      3.14159265358979323846f
#define TILE_ROWS 16
#define NTILES    (NPTS / TILE_ROWS)   // 16384
#define NBLOCKS   1024

// ---------------------------------------------------------------------------
// Kernel 1: fold rotation rows + wavelet scalars into 8 floats per wavelet:
//   q[0..2] = Rx_row, q[3] = 2*dil*OMEGA_0*omega
//   q[4..6] = Rz_row, q[7] = 4*dil*dil*SIGMA_0*sigma^2
// act(p,w) = sin(q3 * (p.Rx) * inv) * exp(-q7 * (1-z) * inv), inv=1/(1e-6+1+z)
// ---------------------------------------------------------------------------
__global__ void gabor_precompute_consts(const float* __restrict__ omega,
                                        const float* __restrict__ sigma,
                                        const float* __restrict__ dilate,
                                        const float* __restrict__ u,
                                        const float* __restrict__ v,
                                        const float* __restrict__ w,
                                        float* __restrict__ consts) {
    int t = threadIdx.x;
    if (t >= WD) return;
    float alpha = 2.0f * PI_F * u[t];
    float cv    = 2.0f * v[t] - 1.0f;
    cv = fminf(fmaxf(cv, -1.0f + 1e-6f), 1.0f - 1e-6f);
    float beta  = acosf(cv);
    float gamma = 2.0f * PI_F * w[t];
    float ca = __cosf(alpha), sa = __sinf(alpha);
    float cb = __cosf(beta),  sb = __sinf(beta);
    float cg = __cosf(gamma), sg = __sinf(gamma);
    // R = Rz(gamma) * Rx(beta) * Rz(alpha); Rx_row = R[0,:], Rz_row = R[2,:]
    float rx0 =  cg * ca - sg * cb * sa;
    float rx1 = -cg * sa - sg * cb * ca;
    float rx2 =  sg * sb;
    float rz0 =  sb * sa;
    float rz1 =  sb * ca;
    float rz2 =  cb;
    float dil = __expf(dilate[t]);
    float cf  = 2.0f * dil * OMEGA_0 * omega[t];
    float cgs = 4.0f * dil * dil * SIGMA_0 * sigma[t] * sigma[t];
    float* o = consts + t * 8;
    o[0] = rx0; o[1] = rx1; o[2] = rx2; o[3] = cf;
    o[4] = rz0; o[5] = rz1; o[6] = rz2; o[7] = cgs;
}

// ---------------------------------------------------------------------------
// Kernel 2: pre-swizzle W_lin (256x128 f32, row-major [out][k]) into f16 WMMA
// B fragments. B is K x N (128 x 256): B[k][n] = W_lin[n][k].
// Fragment unit = (ntile, kchunk): per lane 16 f16.
//   col  = nt*16 + (lane&15); khalf = lane>>4
//   elem i<8 -> k = c*32 + khalf*8 + i ; elem i>=8 -> k = c*32 + 16 + khalf*8 + (i-8)
// Stored at bfrag[ ((nt*4 + c)*32 + lane)*16 + i ]
// ---------------------------------------------------------------------------
__global__ void gabor_precompute_bfrag(const float* __restrict__ W_lin,
                                       _Float16* __restrict__ bfrag) {
    int g = blockIdx.x * blockDim.x + threadIdx.x;   // 0 .. 2047
    if (g >= 16 * 4 * 32) return;
    int lane  = g & 31;
    int c     = (g >> 5) & 3;
    int nt    = g >> 7;
    int col   = nt * 16 + (lane & 15);
    int khalf = lane >> 4;
    _Float16* o = bfrag + (size_t)g * 16;
#pragma unroll
    for (int i = 0; i < 16; ++i) {
        int k = c * 32 + khalf * 8 + (i & 7) + ((i >> 3) << 4);
        o[i] = (_Float16)W_lin[col * WD + k];
    }
}

// ---------------------------------------------------------------------------
// Main fused kernel: 256 threads = 8 waves. Block grid-strides over 16-row
// tiles. Per tile:
//   1) 16 threads stage the 16 points (x,y,z) into LDS (+ prefetch next tile)
//   2) thread t computes activations for row m=t&15, wavelets (t>>4)*8..+7,
//      writes 8 f16 = one b128 into the A-fragment LDS layout
//   3) wave w loads ALL 4 A chunks (8 clause-able ds_load_b128, one wait),
//      then issues 8 back-to-back v_wmma_f32_16x16x32_f16 against its two
//      register-resident B fragments
//   4) bias + ReLU + coalesced NON-TEMPORAL f32 stores (268 MB stream, never
//      re-read: keep it out of L2 so B frags/consts stay resident)
// A-fragment layout (16-bit A 16x32, per ISA): lane = m + 16*khalf,
//   elem i<8 -> K = khalf*8 + i ; elem i>=8 -> K = 16 + khalf*8 + (i-8)
// Thread's 8 wavelets share c, khalf and map to 8 consecutive elems -> b128.
// ---------------------------------------------------------------------------
struct BU { v4f f[2]; };

__global__ __launch_bounds__(256)
void gabor_main(const float* __restrict__ pts_g,
                const float* __restrict__ consts_g,
                const _Float16* __restrict__ bfrag_g,
                const float* __restrict__ b_lin,
                float* __restrict__ out) {
    __shared__ _Float16 s_afrag[4 * 32 * 16];   // 4 KB: [chunk][lane][elem]
    __shared__ float    s_pts[TILE_ROWS * 3];

    const int t    = threadIdx.x;
    const int lane = t & 31;
    const int wave = t >> 5;

    // --- per-thread wavelet constants (fixed across tiles) -> registers ---
    const int m     = t & 15;
    const int wbase = (t >> 4) * 8;             // 8 consecutive wavelets
    float qc[8][8];
#pragma unroll
    for (int j = 0; j < 8; ++j) {
        const float* q = consts_g + (wbase + j) * 8;
#pragma unroll
        for (int e = 0; e < 8; ++e) qc[j][e] = q[e];
    }
    // A-fragment store slot for this thread (8 halfs, 16B aligned)
    const int c_mine  = wbase >> 5;
    const int khalf_m = (wbase >> 3) & 1;
    const int ifix    = (wbase & 16) >> 1;      // 0 or 8
    _Float16* a_store = s_afrag + ((c_mine * 32) + (m + 16 * khalf_m)) * 16 + ifix;

    // --- load this wave's two B fragments (N-tiles 2*wave, 2*wave+1) ---
    union BH { BU u; v16h h; };
    BH b[2][4];
#pragma unroll
    for (int s = 0; s < 2; ++s) {
        int nt = wave * 2 + s;
#pragma unroll
        for (int c = 0; c < 4; ++c) {
            const v4f* p = (const v4f*)(bfrag_g + (size_t)(((nt * 4 + c) * 32) + lane) * 16);
            b[s][c].u.f[0] = p[0];
            b[s][c].u.f[1] = p[1];
        }
    }
    const int   colIn = lane & 15;
    const float bias0 = b_lin[wave * 32 + colIn];
    const float bias1 = b_lin[wave * 32 + 16 + colIn];

    for (int tile = blockIdx.x; tile < NTILES; tile += (int)gridDim.x) {
        const int row0 = tile * TILE_ROWS;

        __syncthreads();                        // s_afrag/s_pts reuse guard
        if (t < TILE_ROWS) {
            const float* pr = pts_g + (size_t)(row0 + t) * 3;
            s_pts[t * 3 + 0] = pr[0];
            s_pts[t * 3 + 1] = pr[1];
            s_pts[t * 3 + 2] = pr[2];
            // prefetch next tile's points (global_prefetch)
            int nrow = row0 + TILE_ROWS * (int)gridDim.x + t;
            if (nrow < NPTS)
                __builtin_prefetch(pts_g + (size_t)nrow * 3, 0, 1);
        }
        __syncthreads();

        const float px = s_pts[m * 3 + 0];
        const float py = s_pts[m * 3 + 1];
        const float pz = s_pts[m * 3 + 2];

        union AV { v4f f; _Float16 h[8]; } av;
#pragma unroll
        for (int j = 0; j < 8; ++j) {
            float x   = px * qc[j][0] + py * qc[j][1] + pz * qc[j][2];
            float z   = px * qc[j][4] + py * qc[j][5] + pz * qc[j][6];
            float inv = __builtin_amdgcn_rcpf(z + 1.000001f);
            float s   = __sinf(qc[j][3] * x * inv);
            float e   = __expf(-qc[j][7] * (1.0f - z) * inv);
            av.h[j]   = (_Float16)(s * e);
        }
        *(v4f*)a_store = av.f;                  // one ds_store_b128
        __syncthreads();

        // --- load all A chunks first (single dscnt wait), then 8 WMMAs ---
        BH a[4];
#pragma unroll
        for (int c = 0; c < 4; ++c) {
            const v4f* ap = (const v4f*)(s_afrag + (size_t)((c * 32) + lane) * 16);
            a[c].u.f[0] = ap[0];
            a[c].u.f[1] = ap[1];
        }
        v8f acc0 = {};
        v8f acc1 = {};
#pragma unroll
        for (int c = 0; c < 4; ++c) {
            acc0 = __builtin_amdgcn_wmma_f32_16x16x32_f16(
                false, a[c].h, false, b[0][c].h, (short)0, acc0, false, false);
            acc1 = __builtin_amdgcn_wmma_f32_16x16x32_f16(
                false, a[c].h, false, b[1][c].h, (short)0, acc1, false, false);
        }

        // --- epilogue: bias + ReLU, coalesced non-temporal stores ---
        const int rbase = row0 + 8 * (lane >> 4);
        float* o0 = out + (size_t)rbase * OUTF + wave * 32 + colIn;
        float* o1 = o0 + 16;
#pragma unroll
        for (int r = 0; r < 8; ++r) {
            __builtin_nontemporal_store(fmaxf(acc0[r] + bias0, 0.0f),
                                        o0 + (size_t)r * OUTF);
            __builtin_nontemporal_store(fmaxf(acc1[r] + bias1, 0.0f),
                                        o1 + (size_t)r * OUTF);
        }
    }
}

// ---------------------------------------------------------------------------
// Launch
// Inputs (setup_inputs order): 0 input(N,3) 1 omega(128) 2 sigma(128)
// 3 dilate(128) 4 u(128) 5 v(128) 6 w(128) 7 W_lin(256,128) 8 b_lin(256)
// ---------------------------------------------------------------------------
extern "C" void kernel_launch(void* const* d_in, const int* in_sizes, int n_in,
                              void* d_out, int out_size, void* d_ws, size_t ws_size,
                              hipStream_t stream) {
    const float* pts    = (const float*)d_in[0];
    const float* omega  = (const float*)d_in[1];
    const float* sigma  = (const float*)d_in[2];
    const float* dilate = (const float*)d_in[3];
    const float* u      = (const float*)d_in[4];
    const float* v      = (const float*)d_in[5];
    const float* w      = (const float*)d_in[6];
    const float* W_lin  = (const float*)d_in[7];
    const float* b_lin  = (const float*)d_in[8];
    float* out = (float*)d_out;

    float*    d_consts = (float*)d_ws;                       // 128*8*4   = 4 KB
    _Float16* d_bfrag  = (_Float16*)((char*)d_ws + 4096);    // 2048*16*2 = 64 KB

    gabor_precompute_consts<<<1, 128, 0, stream>>>(omega, sigma, dilate, u, v, w,
                                                   d_consts);
    gabor_precompute_bfrag<<<8, 256, 0, stream>>>(W_lin, d_bfrag);
    gabor_main<<<NBLOCKS, 256, 0, stream>>>(pts, d_consts, d_bfrag, b_lin, out);
}
